// RealGPT_9981503996508
// MI455X (gfx1250) — matrix operand order repaired and checked by reading, not compiled, not run on hardware
//
#include <hip/hip_runtime.h>

// ---------------------------------------------------------------------------
// GPT-2-like forward (B=2,S=1024,D=1024,H=16,L=6,F=4096,V=50257) for gfx1250.
// All GEMMs run on v_wmma_f32_16x16x32_bf16 (fp32 accumulate); tiles stream
// into LDS via global_load_async_to_lds_b128 with a 2-stage double buffer
// (ASYNCcnt-tracked). Residual stream stays fp32; weights converted+transposed
// to bf16 [N,K] once per call.
// ---------------------------------------------------------------------------

typedef __attribute__((ext_vector_type(16))) __bf16 v16bf;
typedef __attribute__((ext_vector_type(8)))  float  v8f;

#define BM 128
#define BN 128
#define BK 32
#define BKP 40   // padded LDS pitch (bf16 elems) -> 80B row pitch, 16B-aligned

enum {
  MODE_BF16       = 0,  // out bf16 [M,N] (+bias if non-null)
  MODE_BF16_GELU  = 1,  // out bf16, exact GELU(acc+bias)
  MODE_F32_RES    = 2,  // out fp32 [M,N] += acc + bias   (residual stream)
  MODE_F32_SCORES = 3,  // out fp32, acc*scale, causal mask n>m -> -inf
  MODE_F32_BIAS   = 4   // out fp32 [M,N] = acc + bias    (logits)
};

// ---- CDNA5 async copy: global -> LDS, tracked by ASYNCcnt ------------------
__device__ __forceinline__ void async_copy_b128(unsigned ldsByteAddr, const void* gptr) {
  asm volatile("global_load_async_to_lds_b128 %0, %1, off"
               :: "v"(ldsByteAddr), "v"(gptr)
               : "memory");
}
__device__ __forceinline__ unsigned lds_addr(const void* p) {
  return (unsigned)(unsigned long long)p;  // low 32 bits of flat LDS ptr = LDS offset
}

// ---- WMMA fragment gathers from LDS (ISA 7.12.2 bf16 layouts) --------------
__device__ __forceinline__ v16bf load_a_frag(const __bf16* As, int mBase, int lane) {
  union { v16bf v; unsigned u[8]; } r;
  const int half = lane >> 4;                  // 0: K 0..7/16..23, 1: K 8..15/24..31
  const __bf16* row = As + (mBase + (lane & 15)) * BKP;
  #pragma unroll
  for (int i = 0; i < 8; ++i) {
    int k = ((i >> 2) << 4) + (half << 3) + ((i & 3) << 1);
    r.u[i] = *(const unsigned*)(row + k);      // 2 consecutive bf16 (K pair)
  }
  return r.v;
}

__device__ __forceinline__ v16bf load_b_frag(const __bf16* Bs, int nBase, int lane) {
  union { v16bf v; unsigned u[8]; } r;
  const int kh = (lane >> 4) << 3;             // lanes 0-15: K 0..7/16..23, 16-31: K 8..15/24..31
  const __bf16* row = Bs + (nBase + (lane & 15)) * BKP + kh;
  #pragma unroll
  for (int i = 0; i < 8; ++i)
    r.u[i] = *(const unsigned*)(row + ((i >> 2) << 4) + ((i & 3) << 1));
  return r.v;
}

// ---- WMMA GEMM: C[M,N] = A[M,K](bf16) * BT[N,K](bf16)^T, fp32 accum --------
template <int MODE>
__global__ __launch_bounds__(256) void gemm_bf16_wmma(
    const __bf16* __restrict__ A,  int lda, long long strideA,
    const __bf16* __restrict__ BT, int ldb, long long strideB,
    const float*  __restrict__ bias,
    void* __restrict__ Out, int ldo, long long strideO,
    int M, int N, int K, float scale, int causalK)
{
  const int z  = blockIdx.z;
  const int m0 = blockIdx.y * BM;
  const int n0 = blockIdx.x * BN;
  if (MODE == MODE_F32_SCORES && n0 >= m0 + BM) return;  // fully masked tile

  const __bf16* Ab = A  + (long long)z * strideA;
  const __bf16* Bb = BT + (long long)z * strideB;

  __shared__ __bf16 As[2][BM * BKP];
  __shared__ __bf16 Bs[2][BN * BKP];

  const int tid  = threadIdx.x;
  const int lane = tid & 31;
  const int w    = tid >> 5;          // 8 waves
  const int wm   = (w & 1) * 64;      // wave row offset (2 rows of 64)
  const int wn   = (w >> 1) * 32;     // wave col offset (4 cols of 32)

  v8f acc[4][2] = {};

  int kEnd = K;
  if (causalK) { int lim = m0 + BM; kEnd = (K < lim) ? K : lim; }
  const int nk = kEnd / BK;

  const int ldRow = tid >> 2;         // 0..63
  const int ldCol = (tid & 3) << 3;   // 0,8,16,24 (bf16 elems, 16B groups)

  const bool fullTile = (m0 + BM <= M) && (n0 + BN <= N);

  auto computeStage = [&](const __bf16* Ast, const __bf16* Bst) {
    v16bf bfr[2];
    #pragma unroll
    for (int j = 0; j < 2; ++j) bfr[j] = load_b_frag(Bst, wn + j * 16, lane);
    #pragma unroll
    for (int i = 0; i < 4; ++i) {
      v16bf afr = load_a_frag(Ast, wm + i * 16, lane);
      #pragma unroll
      for (int j = 0; j < 2; ++j)
        acc[i][j] = __builtin_amdgcn_wmma_f32_16x16x32_bf16(
            false, afr, false, bfr[j], (short)0, acc[i][j], false, false);
    }
  };

  if (fullTile) {
    // ---- async double-buffered pipeline: 4 async b128 per thread per stage
    auto issueStage = [&](int kt) {
      const int k0 = kt * BK;
      __bf16* Ad = As[kt & 1];
      __bf16* Bd = Bs[kt & 1];
      #pragma unroll
      for (int i = 0; i < 2; ++i) {
        int r = ldRow + i * 64;
        async_copy_b128(lds_addr(&Ad[r * BKP + ldCol]),
                        Ab + (long long)(m0 + r) * lda + k0 + ldCol);
      }
      #pragma unroll
      for (int i = 0; i < 2; ++i) {
        int r = ldRow + i * 64;
        async_copy_b128(lds_addr(&Bd[r * BKP + ldCol]),
                        Bb + (long long)(n0 + r) * ldb + k0 + ldCol);
      }
    };

    issueStage(0);
    for (int kt = 0; kt < nk; ++kt) {
      if (kt + 1 < nk) {
        issueStage(kt + 1);
        asm volatile("s_wait_asynccnt 0x4" ::: "memory");  // stage kt complete
      } else {
        asm volatile("s_wait_asynccnt 0x0" ::: "memory");
      }
      __syncthreads();                 // all waves' stage-kt LDS data visible
      computeStage(As[kt & 1], Bs[kt & 1]);
      __syncthreads();                 // protect buffer before restage
    }
  } else {
    // ---- guarded synchronous path for edge tiles (N edges only here)
    for (int kt = 0; kt < nk; ++kt) {
      const int k0 = kt * BK;
      __syncthreads();
      #pragma unroll
      for (int i = 0; i < 2; ++i) {
        int m = m0 + ldRow + i * 64;
        uint4 val = {0u, 0u, 0u, 0u};
        if (m < M) val = *(const uint4*)(Ab + (long long)m * lda + k0 + ldCol);
        *(uint4*)(&As[0][(ldRow + i * 64) * BKP + ldCol]) = val;
      }
      #pragma unroll
      for (int i = 0; i < 2; ++i) {
        int n = n0 + ldRow + i * 64;
        uint4 val = {0u, 0u, 0u, 0u};
        if (n < N) val = *(const uint4*)(Bb + (long long)n * ldb + k0 + ldCol);
        *(uint4*)(&Bs[0][(ldRow + i * 64) * BKP + ldCol]) = val;
      }
      __syncthreads();
      computeStage(As[0], Bs[0]);
    }
  }

  // ---- store (C/D layout: lane = {halfM, N}, VGPR r = M within half) -------
  const int lcol  = lane & 15;
  const int lrow8 = (lane >> 4) * 8;
  #pragma unroll
  for (int i = 0; i < 4; ++i) {
    #pragma unroll
    for (int j = 0; j < 2; ++j) {
      union { v8f v; float f[8]; } u; u.v = acc[i][j];
      #pragma unroll
      for (int r = 0; r < 8; ++r) {
        int mg = m0 + wm + i * 16 + lrow8 + r;
        int ng = n0 + wn + j * 16 + lcol;
        if (mg >= M || ng >= N) continue;
        float v = u.f[r];
        long long oidx = (long long)z * strideO + (long long)mg * ldo + ng;
        if (MODE == MODE_BF16) {
          if (bias) v += bias[ng];
          ((__bf16*)Out)[oidx] = (__bf16)v;
        } else if (MODE == MODE_BF16_GELU) {
          v += bias[ng];
          v = 0.5f * v * (1.0f + erff(v * 0.70710678118f));
          ((__bf16*)Out)[oidx] = (__bf16)v;
        } else if (MODE == MODE_F32_RES) {
          ((float*)Out)[oidx] += v + bias[ng];
        } else if (MODE == MODE_F32_SCORES) {
          ((float*)Out)[oidx] = (ng > mg) ? -INFINITY : v * scale;
        } else { // MODE_F32_BIAS
          ((float*)Out)[oidx] = v + bias[ng];
        }
      }
    }
  }
}

// ---- wave32 reductions -----------------------------------------------------
__device__ __forceinline__ float wave_sum(float v) {
  #pragma unroll
  for (int o = 16; o > 0; o >>= 1) v += __shfl_xor(v, o, 32);
  return v;
}
__device__ __forceinline__ float wave_max(float v) {
  #pragma unroll
  for (int o = 16; o > 0; o >>= 1) v = fmaxf(v, __shfl_xor(v, o, 32));
  return v;
}

// ---- embedding: x = tok_emb[id] + pos_emb ---------------------------------
__global__ __launch_bounds__(256) void embed_kernel(
    const int* __restrict__ ids, const float* __restrict__ tok,
    const float* __restrict__ pos, float* __restrict__ x, int S, int D)
{
  int row = blockIdx.x;                 // 0..B*S-1
  int token = ids[row];
  int s = row % S;
  int d = threadIdx.x * 4;
  float4 a = *(const float4*)(tok + (long long)token * D + d);
  float4 p = *(const float4*)(pos + (long long)s * D + d);
  float4 r; r.x = a.x + p.x; r.y = a.y + p.y; r.z = a.z + p.z; r.w = a.w + p.w;
  *(float4*)(x + (long long)row * D + d) = r;
}

// ---- LayerNorm: fp32 in -> bf16 out (one block per row, D=1024) ------------
__global__ __launch_bounds__(256) void ln_kernel(
    const float* __restrict__ x, const float* __restrict__ g,
    const float* __restrict__ b, __bf16* __restrict__ out, int D)
{
  __shared__ float sm1[8], sm2[8];
  long long base = (long long)blockIdx.x * D;
  int d = threadIdx.x * 4;
  float4 v = *(const float4*)(x + base + d);
  float s1 = v.x + v.y + v.z + v.w;
  float s2 = v.x * v.x + v.y * v.y + v.z * v.z + v.w * v.w;
  s1 = wave_sum(s1); s2 = wave_sum(s2);
  int lane = threadIdx.x & 31, w = threadIdx.x >> 5;
  if (lane == 0) { sm1[w] = s1; sm2[w] = s2; }
  __syncthreads();
  float t1 = 0.f, t2 = 0.f;
  #pragma unroll
  for (int i = 0; i < 8; ++i) { t1 += sm1[i]; t2 += sm2[i]; }
  float mean = t1 / D;
  float inv  = rsqrtf(t2 / D - mean * mean + 1e-5f);
  float4 gg = *(const float4*)(g + d);
  float4 bb = *(const float4*)(b + d);
  out[base + d + 0] = (__bf16)((v.x - mean) * inv * gg.x + bb.x);
  out[base + d + 1] = (__bf16)((v.y - mean) * inv * gg.y + bb.y);
  out[base + d + 2] = (__bf16)((v.z - mean) * inv * gg.z + bb.z);
  out[base + d + 3] = (__bf16)((v.w - mean) * inv * gg.w + bb.w);
}

// ---- causal softmax: fp32 scores row -> bf16 attn row (zeros past diag) ----
__global__ __launch_bounds__(256) void softmax_kernel(
    const float* __restrict__ sc, __bf16* __restrict__ attn, int S)
{
  __shared__ float sm[8];
  long long base = (long long)blockIdx.x * S;
  int i = blockIdx.x % S;
  int n = i + 1;
  int lane = threadIdx.x & 31, w = threadIdx.x >> 5;

  float mx = -INFINITY;
  for (int j = threadIdx.x; j < n; j += 256) mx = fmaxf(mx, sc[base + j]);
  mx = wave_max(mx);
  if (lane == 0) sm[w] = mx;
  __syncthreads();
  float m2 = -INFINITY;
  #pragma unroll
  for (int k = 0; k < 8; ++k) m2 = fmaxf(m2, sm[k]);
  __syncthreads();

  float s = 0.f;
  for (int j = threadIdx.x; j < n; j += 256) s += __expf(sc[base + j] - m2);
  s = wave_sum(s);
  if (lane == 0) sm[w] = s;
  __syncthreads();
  float tot = 0.f;
  #pragma unroll
  for (int k = 0; k < 8; ++k) tot += sm[k];
  float invs = 1.0f / tot;

  for (int j = threadIdx.x; j < S; j += 256) {
    float o = (j < n) ? __expf(sc[base + j] - m2) * invs : 0.0f;
    attn[base + j] = (__bf16)o;
  }
}

// ---- head split/merge ------------------------------------------------------
__global__ __launch_bounds__(256) void split_heads_kernel(
    const __bf16* __restrict__ q, const __bf16* __restrict__ k,
    const __bf16* __restrict__ v, __bf16* __restrict__ qh,
    __bf16* __restrict__ kh, __bf16* __restrict__ vT,
    int B, int S, int H, int DK)
{
  long long t = (long long)blockIdx.x * 256 + threadIdx.x;
  long long total = (long long)B * S * H * DK;
  if (t >= total) return;
  int dk = t % DK; long long r = t / DK;
  int h = r % H; r /= H;
  int s = r % S; int b = (int)(r / S);
  long long src = (((long long)b * S + s) * H + h) * DK + dk;
  long long dst = (((long long)b * H + h) * S + s) * DK + dk;
  qh[dst] = q[src];
  kh[dst] = k[src];
  vT[(((long long)b * H + h) * DK + dk) * S + s] = v[src];
}

__global__ __launch_bounds__(256) void merge_heads_kernel(
    const __bf16* __restrict__ ctxh, __bf16* __restrict__ ctx,
    int B, int S, int H, int DK)
{
  long long t = (long long)blockIdx.x * 256 + threadIdx.x;
  long long total = (long long)B * S * H * DK;
  if (t >= total) return;
  int dk = t % DK; long long r = t / DK;
  int h = r % H; r /= H;
  int s = r % S; int b = (int)(r / S);
  ctx[(((long long)b * S + s) * H + h) * DK + dk] =
      ctxh[(((long long)b * H + h) * S + s) * DK + dk];
}

// ---- weight convert + transpose: fp32 W[K,N] -> bf16 WT[N,K] ---------------
__global__ __launch_bounds__(256) void conv_transpose_kernel(
    const float* __restrict__ W, __bf16* __restrict__ WT,
    int K, int N, long long wStride, long long wtStride)
{
  __shared__ float tile[32][33];
  int z  = blockIdx.z;
  int n0 = blockIdx.x * 32;
  int k0 = blockIdx.y * 32;
  int tx = threadIdx.x & 31, ty = threadIdx.x >> 5;
  const float* Wb = W + (long long)z * wStride;
  __bf16* WTb = WT + (long long)z * wtStride;
  #pragma unroll
  for (int r = 0; r < 4; ++r) {
    int k = k0 + ty + r * 8;
    if (k < K && (n0 + tx) < N)
      tile[ty + r * 8][tx] = Wb[(long long)k * N + n0 + tx];
  }
  __syncthreads();
  #pragma unroll
  for (int r = 0; r < 4; ++r) {
    int n = n0 + ty + r * 8;
    if (n < N && (k0 + tx) < K)
      WTb[(long long)n * K + k0 + tx] = (__bf16)tile[tx][ty + r * 8];
  }
}

// ---------------------------------------------------------------------------
extern "C" void kernel_launch(void* const* d_in, const int* in_sizes, int n_in,
                              void* d_out, int out_size, void* d_ws, size_t ws_size,
                              hipStream_t stream)
{
  (void)in_sizes; (void)n_in; (void)out_size; (void)ws_size;
  const int  Bc = 2, Sc = 1024, Dc = 1024, Hc = 16, Lc = 6, Fc = 4096, Vc = 50257, DKc = 64;
  const int  Mr = Bc * Sc;                       // 2048 token rows
  const long long DD = (long long)Dc * Dc;
  const long long DF = (long long)Dc * Fc;

  const int*   ids  = (const int*)  d_in[0];
  const float* tok  = (const float*)d_in[1];
  const float* pos  = (const float*)d_in[2];
  const float* Wq   = (const float*)d_in[3];  const float* bq   = (const float*)d_in[4];
  const float* Wk   = (const float*)d_in[5];  const float* bk   = (const float*)d_in[6];
  const float* Wv   = (const float*)d_in[7];  const float* bv   = (const float*)d_in[8];
  const float* Wo   = (const float*)d_in[9];  const float* bo   = (const float*)d_in[10];
  const float* ln1g = (const float*)d_in[11]; const float* ln1b = (const float*)d_in[12];
  const float* fc1w = (const float*)d_in[13]; const float* fc1b = (const float*)d_in[14];
  const float* fc2w = (const float*)d_in[15]; const float* fc2b = (const float*)d_in[16];
  const float* ln2g = (const float*)d_in[17]; const float* ln2b = (const float*)d_in[18];
  const float* lnfg = (const float*)d_in[19]; const float* lnfb = (const float*)d_in[20];
  const float* hw   = (const float*)d_in[21]; const float* hb   = (const float*)d_in[22];

  size_t off = 0;
  auto alloc = [&](size_t bytes) -> void* {
    off = (off + 255) & ~(size_t)255;
    void* p = (char*)d_ws + off;
    off += bytes;
    return p;
  };

  float*  x      = (float*) alloc(sizeof(float) * (size_t)Mr * Dc);
  __bf16* h      = (__bf16*)alloc(sizeof(__bf16) * (size_t)Mr * Dc);
  __bf16* qb     = (__bf16*)alloc(sizeof(__bf16) * (size_t)Mr * Dc);
  __bf16* kb     = (__bf16*)alloc(sizeof(__bf16) * (size_t)Mr * Dc);
  __bf16* vb     = (__bf16*)alloc(sizeof(__bf16) * (size_t)Mr * Dc);
  __bf16* qh     = (__bf16*)alloc(sizeof(__bf16) * (size_t)Mr * Dc);
  __bf16* kh     = (__bf16*)alloc(sizeof(__bf16) * (size_t)Mr * Dc);
  __bf16* vT     = (__bf16*)alloc(sizeof(__bf16) * (size_t)Mr * Dc);
  __bf16* ctxh   = (__bf16*)alloc(sizeof(__bf16) * (size_t)Mr * Dc);
  __bf16* ctx    = (__bf16*)alloc(sizeof(__bf16) * (size_t)Mr * Dc);
  float*  scores = (float*) alloc(sizeof(float) * (size_t)Bc * Hc * Sc * Sc);
  __bf16* attn   = (__bf16*)alloc(sizeof(__bf16) * (size_t)Bc * Hc * Sc * Sc);
  __bf16* ffh    = (__bf16*)alloc(sizeof(__bf16) * (size_t)Mr * Fc);
  __bf16* WqT    = (__bf16*)alloc(sizeof(__bf16) * (size_t)Lc * DD);
  __bf16* WkT    = (__bf16*)alloc(sizeof(__bf16) * (size_t)Lc * DD);
  __bf16* WvT    = (__bf16*)alloc(sizeof(__bf16) * (size_t)Lc * DD);
  __bf16* WoT    = (__bf16*)alloc(sizeof(__bf16) * (size_t)Lc * DD);
  __bf16* fc1T   = (__bf16*)alloc(sizeof(__bf16) * (size_t)Lc * DF);
  __bf16* fc2T   = (__bf16*)alloc(sizeof(__bf16) * (size_t)Lc * DF);
  __bf16* headT  = (__bf16*)alloc(sizeof(__bf16) * (size_t)Vc * Dc);

  dim3 t256(256);

  // 1) weight convert + transpose (one-time per call)
  conv_transpose_kernel<<<dim3(Dc / 32, Dc / 32, Lc), t256, 0, stream>>>(Wq, WqT, Dc, Dc, DD, DD);
  conv_transpose_kernel<<<dim3(Dc / 32, Dc / 32, Lc), t256, 0, stream>>>(Wk, WkT, Dc, Dc, DD, DD);
  conv_transpose_kernel<<<dim3(Dc / 32, Dc / 32, Lc), t256, 0, stream>>>(Wv, WvT, Dc, Dc, DD, DD);
  conv_transpose_kernel<<<dim3(Dc / 32, Dc / 32, Lc), t256, 0, stream>>>(Wo, WoT, Dc, Dc, DD, DD);
  conv_transpose_kernel<<<dim3(Fc / 32, Dc / 32, Lc), t256, 0, stream>>>(fc1w, fc1T, Dc, Fc, DF, DF);
  conv_transpose_kernel<<<dim3(Dc / 32, Fc / 32, Lc), t256, 0, stream>>>(fc2w, fc2T, Fc, Dc, DF, DF);
  conv_transpose_kernel<<<dim3((Vc + 31) / 32, Dc / 32, 1), t256, 0, stream>>>(hw, headT, Dc, Vc, 0, 0);

  // 2) embeddings
  embed_kernel<<<Mr, t256, 0, stream>>>(ids, tok, pos, x, Sc, Dc);

  const dim3 gD (Dc / BN, Mr / BM, 1);          // [2048 x 1024] projections
  const dim3 gF (Fc / BN, Mr / BM, 1);          // [2048 x 4096] fc1
  const dim3 gSc(Sc / BN, Sc / BM, Bc * Hc);    // scores per (b,h)
  const dim3 gCt(1,       Sc / BM, Bc * Hc);    // ctx per (b,h), N=64
  const long long sQH = (long long)Sc * DKc;
  const long long sSS = (long long)Sc * Sc;
  const int nElem = Mr * Dc;                    // 2M
  const int nBlkE = (nElem + 255) / 256;

  for (int l = 0; l < Lc; ++l) {
    ln_kernel<<<Mr, t256, 0, stream>>>(x, ln1g + (size_t)l * Dc, ln1b + (size_t)l * Dc, h, Dc);

    gemm_bf16_wmma<MODE_BF16><<<gD, t256, 0, stream>>>(
        h, Dc, 0, WqT + (size_t)l * DD, Dc, 0, bq + (size_t)l * Dc,
        qb, Dc, 0, Mr, Dc, Dc, 1.0f, 0);
    gemm_bf16_wmma<MODE_BF16><<<gD, t256, 0, stream>>>(
        h, Dc, 0, WkT + (size_t)l * DD, Dc, 0, bk + (size_t)l * Dc,
        kb, Dc, 0, Mr, Dc, Dc, 1.0f, 0);
    gemm_bf16_wmma<MODE_BF16><<<gD, t256, 0, stream>>>(
        h, Dc, 0, WvT + (size_t)l * DD, Dc, 0, bv + (size_t)l * Dc,
        vb, Dc, 0, Mr, Dc, Dc, 1.0f, 0);

    split_heads_kernel<<<nBlkE, t256, 0, stream>>>(qb, kb, vb, qh, kh, vT, Bc, Sc, Hc, DKc);

    // scores = (q @ k^T) / sqrt(DK), causal
    gemm_bf16_wmma<MODE_F32_SCORES><<<gSc, t256, 0, stream>>>(
        qh, DKc, sQH, kh, DKc, sQH, nullptr,
        scores, Sc, sSS, Sc, Sc, DKc, 0.125f, 0);

    softmax_kernel<<<Bc * Hc * Sc, t256, 0, stream>>>(scores, attn, Sc);

    // ctx = attn @ v  (K-loop clipped at diagonal)
    gemm_bf16_wmma<MODE_BF16><<<gCt, t256, 0, stream>>>(
        attn, Sc, sSS, vT, Sc, (long long)DKc * Sc, nullptr,
        ctxh, DKc, sQH, Sc, DKc, Sc, 1.0f, 1);

    merge_heads_kernel<<<nBlkE, t256, 0, stream>>>(ctxh, ctx, Bc, Sc, Hc, DKc);

    // x += ctx @ Wo + bo
    gemm_bf16_wmma<MODE_F32_RES><<<gD, t256, 0, stream>>>(
        ctx, Dc, 0, WoT + (size_t)l * DD, Dc, 0, bo + (size_t)l * Dc,
        x, Dc, 0, Mr, Dc, Dc, 1.0f, 0);

    ln_kernel<<<Mr, t256, 0, stream>>>(x, ln2g + (size_t)l * Dc, ln2b + (size_t)l * Dc, h, Dc);

    gemm_bf16_wmma<MODE_BF16_GELU><<<gF, t256, 0, stream>>>(
        h, Dc, 0, fc1T + (size_t)l * DF, Dc, 0, fc1b + (size_t)l * Fc,
        ffh, Fc, 0, Mr, Fc, Dc, 1.0f, 0);

    gemm_bf16_wmma<MODE_F32_RES><<<gD, t256, 0, stream>>>(
        ffh, Fc, 0, fc2T + (size_t)l * DF, Fc, 0, fc2b + (size_t)l * Dc,
        x, Dc, 0, Mr, Dc, Fc, 1.0f, 0);
  }

  // final LN + LM head
  ln_kernel<<<Mr, t256, 0, stream>>>(x, lnfg, lnfb, h, Dc);
  gemm_bf16_wmma<MODE_F32_BIAS><<<dim3((Vc + BN - 1) / BN, Mr / BM, 1), t256, 0, stream>>>(
      h, Dc, 0, headT, Dc, 0, hb, (float*)d_out, Vc, 0, Mr, Vc, Dc, 1.0f, 0);
}
